// SPOT_41592463294749
// MI455X (gfx1250) — matrix operand order repaired and checked by reading, not compiled, not run on hardware
//
#include <hip/hip_runtime.h>
#include <hip/hip_bf16.h>
#include <math.h>

// ---------------------------------------------------------------------------
// Types for CDNA5 WMMA
// ---------------------------------------------------------------------------
typedef __attribute__((ext_vector_type(16))) __bf16 v16bf;
typedef __attribute__((ext_vector_type(8)))  __bf16 v8bf;
typedef __attribute__((ext_vector_type(4)))  __bf16 v4bf;
typedef __attribute__((ext_vector_type(8)))  float  v8f;

// Model constants (from the reference)
#define NF   256
#define D_IN 256
#define HID  64
#define OUTD 16
#define NH   3
#define SLOPE 0.2f

// GEMM tiling
#define BM 64
#define BN 64
#define BK 32
#define LDS_STRIDE 40   // bf16 elems per row: 80 B, 16-B aligned, bank-skewed

// ---------------------------------------------------------------------------
// Helpers
// ---------------------------------------------------------------------------
__device__ __forceinline__ __bf16 f32_to_bf16(float f) {
  unsigned int u = __builtin_bit_cast(unsigned int, f);
  unsigned int r = (u + 0x7FFFu + ((u >> 16) & 1u)) >> 16;   // RNE
  return __builtin_bit_cast(__bf16, (unsigned short)r);
}
__device__ __forceinline__ v4bf pack4(float a, float b, float c, float d) {
  v4bf p;
  p[0] = f32_to_bf16(a); p[1] = f32_to_bf16(b);
  p[2] = f32_to_bf16(c); p[3] = f32_to_bf16(d);
  return p;
}

// order-preserving float<->uint for atomicMax on floats
__device__ __forceinline__ unsigned int f2ord(float f) {
  unsigned int u = __builtin_bit_cast(unsigned int, f);
  return (u & 0x80000000u) ? ~u : (u | 0x80000000u);
}
__device__ __forceinline__ float ord2f(unsigned int u) {
  u = (u & 0x80000000u) ? (u & 0x7FFFFFFFu) : ~u;
  return __builtin_bit_cast(float, u);
}

__device__ __forceinline__ float lrelu(float x) { return x > 0.f ? x : SLOPE * x; }
__device__ __forceinline__ float elu1(float x)  { return x > 0.f ? x : expm1f(x); }

// ---------------------------------------------------------------------------
// Generic GEMM: C[M,Nn] = act(A @ B + bias)
//   A element (m,k) at A[m*a_rs + k*a_cs]   (handles transposed inputs)
//   B element (k,n) at B[k*b_rs + n*b_cs]
// bf16 WMMA, f32 accumulation. Block: 128 threads (4 waves).
// Tile: 64(M) x 64(N), K-step 32. Each wave: one 16-col slice, 4 accumulators.
// LDS: As row-major [64][32], Bs TRANSPOSED [n][k] = [64][32] so every lane's
// fragment is a pair of contiguous 16-B runs -> ds_load_b128.
// Bounds are handled inside the vectorized staging (per-chunk masking), so the
// fast path applies to edge tiles and small-N/small-K GEMMs too.
// ---------------------------------------------------------------------------
__global__ void __launch_bounds__(128)
gemm_bf16_wmma(const float* __restrict__ A, long a_rs, long a_cs,
               const float* __restrict__ B, long b_rs, long b_cs,
               const float* __restrict__ bias,
               float* __restrict__ C, int ldc,
               int M, int Nn, int K, int act)
{
  __shared__ __align__(16) __bf16 As[BM * LDS_STRIDE];
  __shared__ __align__(16) __bf16 Bs[BN * LDS_STRIDE];   // [n][k]

  const int m0   = blockIdx.y * BM;
  const int n0   = blockIdx.x * BN;
  const int tid  = threadIdx.x;
  const int wave = tid >> 5;      // 0..3 -> 16-col subtile
  const int lane = tid & 31;
  const int g    = lane >> 4;     // lane group
  const int lr   = lane & 15;

  v8f acc0 = {}, acc1 = {}, acc2 = {}, acc3 = {};

  for (int k0 = 0; k0 < K; k0 += BK) {
    // ---------------- stage A tile: As[r][k] ----------------
    if (a_cs == 1) {                      // contiguous along k
#pragma unroll
      for (int it = 0; it < 4; ++it) {
        int idx = tid + it * 128;         // 0..511
        int r = idx >> 3, c4 = idx & 7;
        int m = m0 + r, k = k0 + c4 * 4;
        float4 v;
        if (m < M && k + 4 <= K) {
          v = *(const float4*)(A + (long)m * a_rs + k);
        } else {
          v.x = (m < M && k + 0 < K) ? A[(long)m * a_rs + k + 0] : 0.f;
          v.y = (m < M && k + 1 < K) ? A[(long)m * a_rs + k + 1] : 0.f;
          v.z = (m < M && k + 2 < K) ? A[(long)m * a_rs + k + 2] : 0.f;
          v.w = (m < M && k + 3 < K) ? A[(long)m * a_rs + k + 3] : 0.f;
        }
        *(v4bf*)&As[r * LDS_STRIDE + c4 * 4] = pack4(v.x, v.y, v.z, v.w);
      }
    } else if (a_rs == 1) {               // contiguous along m (transposed A)
#pragma unroll
      for (int it = 0; it < 4; ++it) {
        int idx = tid + it * 128;
        int kk = idx >> 4, m4 = idx & 15;
        int k = k0 + kk, m = m0 + m4 * 4;
        bool kok = (k < K);
        float4 v;
        if (kok && m + 4 <= M) {
          v = *(const float4*)(A + (long)k * a_cs + m);
        } else {
          v.x = (kok && m + 0 < M) ? A[(long)k * a_cs + m + 0] : 0.f;
          v.y = (kok && m + 1 < M) ? A[(long)k * a_cs + m + 1] : 0.f;
          v.z = (kok && m + 2 < M) ? A[(long)k * a_cs + m + 2] : 0.f;
          v.w = (kok && m + 3 < M) ? A[(long)k * a_cs + m + 3] : 0.f;
        }
        As[(m4 * 4 + 0) * LDS_STRIDE + kk] = f32_to_bf16(v.x);
        As[(m4 * 4 + 1) * LDS_STRIDE + kk] = f32_to_bf16(v.y);
        As[(m4 * 4 + 2) * LDS_STRIDE + kk] = f32_to_bf16(v.z);
        As[(m4 * 4 + 3) * LDS_STRIDE + kk] = f32_to_bf16(v.w);
      }
    } else {                              // generic strides (unused here)
#pragma unroll
      for (int it = 0; it < 16; ++it) {
        int idx = tid + it * 128;
        int r = idx >> 5, c = idx & 31;
        int m = m0 + r, k = k0 + c;
        float v = (m < M && k < K) ? A[(long)m * a_rs + (long)k * a_cs] : 0.f;
        As[r * LDS_STRIDE + c] = f32_to_bf16(v);
      }
    }

    // ---------------- stage B tile transposed: Bs[n][k] ----------------
    if (b_rs == 1) {                      // contiguous along k (transposed B)
#pragma unroll
      for (int it = 0; it < 4; ++it) {
        int idx = tid + it * 128;
        int nn = idx >> 3, k4 = idx & 7;
        int n = n0 + nn, k = k0 + k4 * 4;
        bool nok = (n < Nn);
        float4 v;
        if (nok && k + 4 <= K) {
          v = *(const float4*)(B + (long)n * b_cs + k);
        } else {
          v.x = (nok && k + 0 < K) ? B[(long)n * b_cs + k + 0] : 0.f;
          v.y = (nok && k + 1 < K) ? B[(long)n * b_cs + k + 1] : 0.f;
          v.z = (nok && k + 2 < K) ? B[(long)n * b_cs + k + 2] : 0.f;
          v.w = (nok && k + 3 < K) ? B[(long)n * b_cs + k + 3] : 0.f;
        }
        *(v4bf*)&Bs[nn * LDS_STRIDE + k4 * 4] = pack4(v.x, v.y, v.z, v.w);
      }
    } else if (b_cs == 1) {               // contiguous along n (row-major B)
#pragma unroll
      for (int it = 0; it < 4; ++it) {
        int idx = tid + it * 128;
        int kk = idx >> 4, n4 = idx & 15;
        int k = k0 + kk, n = n0 + n4 * 4;
        bool kok = (k < K);
        float4 v;
        if (kok && n + 4 <= Nn) {
          v = *(const float4*)(B + (long)k * b_rs + n);
        } else {
          v.x = (kok && n + 0 < Nn) ? B[(long)k * b_rs + n + 0] : 0.f;
          v.y = (kok && n + 1 < Nn) ? B[(long)k * b_rs + n + 1] : 0.f;
          v.z = (kok && n + 2 < Nn) ? B[(long)k * b_rs + n + 2] : 0.f;
          v.w = (kok && n + 3 < Nn) ? B[(long)k * b_rs + n + 3] : 0.f;
        }
        Bs[(n4 * 4 + 0) * LDS_STRIDE + kk] = f32_to_bf16(v.x);
        Bs[(n4 * 4 + 1) * LDS_STRIDE + kk] = f32_to_bf16(v.y);
        Bs[(n4 * 4 + 2) * LDS_STRIDE + kk] = f32_to_bf16(v.z);
        Bs[(n4 * 4 + 3) * LDS_STRIDE + kk] = f32_to_bf16(v.w);
      }
    } else {                              // generic strides (unused here)
#pragma unroll
      for (int it = 0; it < 16; ++it) {
        int idx = tid + it * 128;
        int kk = idx >> 6, nn = idx & 63;
        int k = k0 + kk, n = n0 + nn;
        float v = (k < K && n < Nn) ? B[(long)k * b_rs + (long)n * b_cs] : 0.f;
        Bs[nn * LDS_STRIDE + kk] = f32_to_bf16(v);
      }
    }
    __syncthreads();

    // ---- B fragment: lanes 0-15 -> K 0..15, lanes 16-31 -> K 16..31 ----
    const __bf16* bp = &Bs[(wave * 16 + lr) * LDS_STRIDE + g * 16];
    v8bf bLo = *(const v8bf*)bp;
    v8bf bHi = *(const v8bf*)(bp + 8);
    v16bf bFrag = __builtin_shufflevector(bLo, bHi,
        0,1,2,3,4,5,6,7,8,9,10,11,12,13,14,15);

    // ---- 4 A fragments (rows mt*16+lr), K run [16g,16g+15] ----
#pragma unroll
    for (int mt = 0; mt < 4; ++mt) {
      const __bf16* ap = &As[(mt * 16 + lr) * LDS_STRIDE + g * 16];
      v8bf aLo = *(const v8bf*)ap;
      v8bf aHi = *(const v8bf*)(ap + 8);
      v16bf aFrag = __builtin_shufflevector(aLo, aHi,
          0,1,2,3,4,5,6,7,8,9,10,11,12,13,14,15);
      v8f* pacc = (mt == 0) ? &acc0 : (mt == 1) ? &acc1 : (mt == 2) ? &acc2 : &acc3;
      *pacc = __builtin_amdgcn_wmma_f32_16x16x32_bf16(
          false, aFrag, false, bFrag, (short)0, *pacc, false, false);
    }
    __syncthreads();
  }

  // ---- write C: VGPR r -> row r + 8*g within each 16-row m-subtile ----
  const int n = n0 + wave * 16 + lr;
  const float bv = (bias && n < Nn) ? bias[n] : 0.f;
#pragma unroll
  for (int mt = 0; mt < 4; ++mt) {
    v8f a = (mt == 0) ? acc0 : (mt == 1) ? acc1 : (mt == 2) ? acc2 : acc3;
#pragma unroll
    for (int r = 0; r < 8; ++r) {
      int m = m0 + mt * 16 + r + 8 * g;
      if (m < M && n < Nn) {
        float v = a[r] + bv;
        if (act == 1) v = fmaxf(v, 0.f);
        C[(long)m * ldc + n] = v;
      }
    }
  }
}

// ---------------------------------------------------------------------------
// GATv2 edge passes (segment softmax over dst, H=3 heads, C=64 per head)
// ---------------------------------------------------------------------------
__global__ void gat_logits(const float* __restrict__ xl, const float* __restrict__ xr,
                           const float* __restrict__ att,
                           const int* __restrict__ src, const int* __restrict__ dst,
                           float* __restrict__ ebuf, unsigned int* __restrict__ emaxU,
                           int E)
{
  int t = blockIdx.x * blockDim.x + threadIdx.x;
  if (t >= E * NH) return;
  int e = t / NH, h = t - e * NH;
  int s = src[e], d = dst[e];
  const float* pl = xl + (long)s * (NH * HID) + h * HID;
  const float* pr = xr + (long)d * (NH * HID) + h * HID;
  const float* pa = att + h * HID;
  float a = 0.f;
#pragma unroll 8
  for (int c = 0; c < HID; ++c) a += lrelu(pl[c] + pr[c]) * pa[c];
  ebuf[t] = a;
  atomicMax(&emaxU[d * NH + h], f2ord(a));
}

__global__ void emax_decode(const unsigned int* __restrict__ emaxU,
                            float* __restrict__ emax, int n3)
{
  int t = blockIdx.x * blockDim.x + threadIdx.x;
  if (t >= n3) return;
  float f = ord2f(emaxU[t]);
  unsigned int bits = __builtin_bit_cast(unsigned int, f);
  bool finite = (bits & 0x7F800000u) != 0x7F800000u;
  emax[t] = (emaxU[t] != 0u && finite) ? f : 0.f;   // empty segment -> 0
}

__global__ void gat_expsum(const int* __restrict__ dst,
                           float* __restrict__ ebuf,
                           const float* __restrict__ emax,
                           float* __restrict__ dsum, int E)
{
  int t = blockIdx.x * blockDim.x + threadIdx.x;
  if (t >= E * NH) return;
  int e = t / NH, h = t - e * NH;
  int d = dst[e];
  float p = expf(ebuf[t] - emax[d * NH + h]);
  ebuf[t] = p;
  atomicAdd(&dsum[d * NH + h], p);
}

__global__ void gat_scatter(const float* __restrict__ xl,
                            const int* __restrict__ src, const int* __restrict__ dst,
                            const float* __restrict__ ebuf,
                            const float* __restrict__ dsum,
                            float* __restrict__ accum, int E)
{
  long t = (long)blockIdx.x * blockDim.x + threadIdx.x;
  if (t >= (long)E * (NH * HID)) return;
  int e = (int)(t / (NH * HID));
  int j = (int)(t - (long)e * (NH * HID));
  int h = j >> 6;                       // j / HID
  int s = src[e], d = dst[e];
  float denom = dsum[d * NH + h];
  float alpha = ebuf[(long)e * NH + h] / fmaxf(denom, 1e-16f);
  atomicAdd(&accum[(long)d * (NH * HID) + j], alpha * xl[(long)s * (NH * HID) + j]);
}

// head mean + bias + ELU, written into concat destination (stride 2*HID)
__global__ void gat_finalize(const float* __restrict__ accum,
                             const float* __restrict__ b,
                             float* __restrict__ out, int stride, int off, int n)
{
  int t = blockIdx.x * blockDim.x + threadIdx.x;
  if (t >= n * HID) return;
  int row = t >> 6, c = t & (HID - 1);
  const float* a = accum + (long)row * (NH * HID);
  float v = (a[c] + a[HID + c] + a[2 * HID + c]) * (1.f / 3.f) + b[c];
  out[(long)row * stride + off + c] = elu1(v);
}

// ---------------------------------------------------------------------------
// Cross-attention: per row, e = lrelu([Xg|Xs]@a1, [Xg|Xs]@a2), softmax over 2,
// res = lam0*Xg + lam1*Xs
// ---------------------------------------------------------------------------
__global__ void __launch_bounds__(256)
cross_attn(const float* __restrict__ Xg, const float* __restrict__ Xs,
           const float* __restrict__ a1, const float* __restrict__ a2,
           float* __restrict__ res, float* __restrict__ lam)
{
  __shared__ float s1[256], s2[256];
  __shared__ float l0s, l1s;
  int n = blockIdx.x;
  int t = threadIdx.x;
  float gv = Xg[(long)n * NF + t];
  float sv = Xs[(long)n * NF + t];
  s1[t] = gv * a1[t] + sv * a1[NF + t];
  s2[t] = gv * a2[t] + sv * a2[NF + t];
  __syncthreads();
  for (int off = 128; off > 0; off >>= 1) {
    if (t < off) { s1[t] += s1[t + off]; s2[t] += s2[t + off]; }
    __syncthreads();
  }
  if (t == 0) {
    float e1 = lrelu(s1[0]), e2 = lrelu(s2[0]);
    float m  = fmaxf(e1, e2);
    float p0 = expf(e1 - m), p1 = expf(e2 - m);
    float inv = 1.f / (p0 + p1);
    l0s = p0 * inv; l1s = p1 * inv;
    lam[(long)n * 2 + 0] = l0s;
    lam[(long)n * 2 + 1] = l1s;
  }
  __syncthreads();
  res[(long)n * NF + t] = l0s * gv + l1s * sv;
}

// ---------------------------------------------------------------------------
// Host orchestration
// ---------------------------------------------------------------------------
static inline int cdiv(long a, long b) { return (int)((a + b - 1) / b); }

static void launch_gemm(hipStream_t st,
                        const float* A, long a_rs, long a_cs,
                        const float* B, long b_rs, long b_cs,
                        const float* bias, float* C, int ldc,
                        int M, int Nn, int K, int act)
{
  dim3 grid(cdiv(Nn, BN), cdiv(M, BM));
  gemm_bf16_wmma<<<grid, 128, 0, st>>>(A, a_rs, a_cs, B, b_rs, b_cs, bias, C, ldc, M, Nn, K, act);
}

struct GatScratch {
  float* ebuf; unsigned int* emaxU; float* emax; float* dsum; float* accum;
};

static void run_gat(hipStream_t st, int N,
                    const float* xl, const float* xr,
                    const float* att, const float* b,
                    const int* src, const int* dst, int E,
                    GatScratch ws, float* out, int stride, int off)
{
  hipMemsetAsync(ws.emaxU, 0, (size_t)N * NH * sizeof(unsigned int), st);
  hipMemsetAsync(ws.dsum,  0, (size_t)N * NH * sizeof(float), st);
  hipMemsetAsync(ws.accum, 0, (size_t)N * NH * HID * sizeof(float), st);

  gat_logits<<<cdiv((long)E * NH, 256), 256, 0, st>>>(xl, xr, att, src, dst, ws.ebuf, ws.emaxU, E);
  emax_decode<<<cdiv((long)N * NH, 256), 256, 0, st>>>(ws.emaxU, ws.emax, N * NH);
  gat_expsum<<<cdiv((long)E * NH, 256), 256, 0, st>>>(dst, ws.ebuf, ws.emax, ws.dsum, E);
  gat_scatter<<<cdiv((long)E * NH * HID, 256), 256, 0, st>>>(xl, src, dst, ws.ebuf, ws.dsum, ws.accum, E);
  gat_finalize<<<cdiv((long)N * HID, 256), 256, 0, st>>>(ws.accum, b, out, stride, off, N);
}

extern "C" void kernel_launch(void* const* d_in, const int* in_sizes, int n_in,
                              void* d_out, int out_size, void* d_ws, size_t ws_size,
                              hipStream_t stream)
{
  // ---- inputs (setup_inputs dict order) ----
  const float* gene_freq   = (const float*)d_in[0];   // [NF, G]
  const float* gene_eigT   = (const float*)d_in[1];   // [G, N]
  const float* spot_freq   = (const float*)d_in[2];   // [N, D_IN]
  const float* spot_eigT   = (const float*)d_in[3];   // [D_IN, NF]
  const int*   spotnet     = (const int*)d_in[4];     // [2, E1]
  const int*   spotnet_cl  = (const int*)d_in[5];     // [2, E2]

  const int G  = in_sizes[0] / NF;
  const int N  = in_sizes[2] / D_IN;
  const int E1 = in_sizes[4] / 2;
  const int E2 = in_sizes[5] / 2;

  // GAT params: c1(6..11) c1p(12..17) c3(18..23) c3p(24..29)
  const float* c1_Wl  = (const float*)d_in[6];  const float* c1_bl  = (const float*)d_in[7];
  const float* c1_Wr  = (const float*)d_in[8];  const float* c1_br  = (const float*)d_in[9];
  const float* c1_att = (const float*)d_in[10]; const float* c1_b   = (const float*)d_in[11];
  const float* c1p_Wl = (const float*)d_in[12]; const float* c1p_bl = (const float*)d_in[13];
  const float* c1p_Wr = (const float*)d_in[14]; const float* c1p_br = (const float*)d_in[15];
  const float* c1p_att= (const float*)d_in[16]; const float* c1p_b  = (const float*)d_in[17];
  const float* c3_Wl  = (const float*)d_in[18]; const float* c3_bl  = (const float*)d_in[19];
  const float* c3_Wr  = (const float*)d_in[20]; const float* c3_br  = (const float*)d_in[21];
  const float* c3_att = (const float*)d_in[22]; const float* c3_b   = (const float*)d_in[23];
  const float* c3p_Wl = (const float*)d_in[24]; const float* c3p_bl = (const float*)d_in[25];
  const float* c3p_Wr = (const float*)d_in[26]; const float* c3p_br = (const float*)d_in[27];
  const float* c3p_att= (const float*)d_in[28]; const float* c3p_b  = (const float*)d_in[29];
  const float* W2 = (const float*)d_in[30]; const float* b2 = (const float*)d_in[31];
  const float* W4 = (const float*)d_in[32]; const float* b4 = (const float*)d_in[33];
  const float* a1 = (const float*)d_in[34]; const float* a2 = (const float*)d_in[35];

  const int* src1 = spotnet;            const int* dst1 = spotnet + E1;
  const int* src2 = spotnet_cl;         const int* dst2 = spotnet_cl + E2;

  // ---- outputs: res[N,NF] | lam[N,2] | h2[N,OUT] | h4[N,D_IN] (flat) ----
  float* out   = (float*)d_out;
  float* o_res = out;
  float* o_lam = o_res + (long)N * NF;
  float* o_h2  = o_lam + (long)N * 2;
  float* o_h4  = o_h2  + (long)N * OUTD;

  // ---- workspace layout ----
  float* w = (float*)d_ws;
  float* xlA = w;                 w += (long)N * NH * HID;   // 192/row
  float* xrA = w;                 w += (long)N * NH * HID;
  float* xlB = w;                 w += (long)N * NH * HID;
  float* xrB = w;                 w += (long)N * NH * HID;
  float* accum = w;               w += (long)N * NH * HID;
  float* h1  = w;                 w += (long)N * 2 * HID;    // 128/row (also h3)
  float* ebuf = w;                w += (long)E1 * NH;        // E1 >= E2
  unsigned int* emaxU = (unsigned int*)w; w += (long)N * NH;
  float* emax = w;                w += (long)N * NH;
  float* dsum = w;                w += (long)N * NH;
  float* Xg = w;                  w += (long)N * NF;
  float* Xs = w;                  w += (long)N * NF;
  (void)ws_size; (void)n_in; (void)out_size;

  GatScratch gs{ebuf, emaxU, emax, dsum, accum};

  // ---- Layer 1 projections: spot_freq [N,256] @ W[256,192] + bias ----
  launch_gemm(stream, spot_freq, D_IN, 1, c1_Wl,  NH * HID, 1, c1_bl,  xlA, NH * HID, N, NH * HID, D_IN, 0);
  launch_gemm(stream, spot_freq, D_IN, 1, c1_Wr,  NH * HID, 1, c1_br,  xrA, NH * HID, N, NH * HID, D_IN, 0);
  launch_gemm(stream, spot_freq, D_IN, 1, c1p_Wl, NH * HID, 1, c1p_bl, xlB, NH * HID, N, NH * HID, D_IN, 0);
  launch_gemm(stream, spot_freq, D_IN, 1, c1p_Wr, NH * HID, 1, c1p_br, xrB, NH * HID, N, NH * HID, D_IN, 0);

  // ---- Layer 1 GATs -> elu(concat) into h1 [N,128] ----
  run_gat(stream, N, xlA, xrA, c1_att,  c1_b,  src1, dst1, E1, gs, h1, 2 * HID, 0);
  run_gat(stream, N, xlB, xrB, c1p_att, c1p_b, src2, dst2, E2, gs, h1, 2 * HID, HID);

  // ---- h2 = h1 @ W2 + b2 -> d_out slice [N,16] ----
  launch_gemm(stream, h1, 2 * HID, 1, W2, OUTD, 1, b2, o_h2, OUTD, N, OUTD, 2 * HID, 0);

  // ---- Layer 2 projections: h2 [N,16] @ W[16,192] + bias (K=16, zero-padded) ----
  launch_gemm(stream, o_h2, OUTD, 1, c3_Wl,  NH * HID, 1, c3_bl,  xlA, NH * HID, N, NH * HID, OUTD, 0);
  launch_gemm(stream, o_h2, OUTD, 1, c3_Wr,  NH * HID, 1, c3_br,  xrA, NH * HID, N, NH * HID, OUTD, 0);
  launch_gemm(stream, o_h2, OUTD, 1, c3p_Wl, NH * HID, 1, c3p_bl, xlB, NH * HID, N, NH * HID, OUTD, 0);
  launch_gemm(stream, o_h2, OUTD, 1, c3p_Wr, NH * HID, 1, c3p_br, xrB, NH * HID, N, NH * HID, OUTD, 0);

  // ---- Layer 2 GATs -> elu(concat) into h1 (reused as h3) ----
  run_gat(stream, N, xlA, xrA, c3_att,  c3_b,  src1, dst1, E1, gs, h1, 2 * HID, 0);
  run_gat(stream, N, xlB, xrB, c3p_att, c3p_b, src2, dst2, E2, gs, h1, 2 * HID, HID);

  // ---- h4 = h3 @ W4 + b4 -> d_out slice [N,256] ----
  launch_gemm(stream, h1, 2 * HID, 1, W4, D_IN, 1, b4, o_h4, D_IN, N, D_IN, 2 * HID, 0);

  // ---- Xg = relu(gene_eigT^T @ gene_freq^T): [N,G]@[G,NF], both transposed ----
  // A(m,k) = gene_eigT[k*N + m]  -> a_rs=1, a_cs=N  (contiguous along m)
  // B(k,n) = gene_freq[n*G + k]  -> b_rs=1, b_cs=G  (contiguous along k)
  launch_gemm(stream, gene_eigT, 1, N, gene_freq, 1, G, nullptr, Xg, NF, N, NF, G, 1);

  // ---- Xs = relu(h4 @ spot_eigT): [N,256]@[256,256] ----
  launch_gemm(stream, o_h4, D_IN, 1, spot_eigT, NF, 1, nullptr, Xs, NF, N, NF, D_IN, 1);

  // ---- cross attention -> res, lam ----
  cross_attn<<<N, 256, 0, stream>>>(Xg, Xs, a1, a2, o_res, o_lam);
}
